// Attention_23244363006193
// MI455X (gfx1250) — matrix operand order repaired and checked by reading, not compiled, not run on hardware
//
#include <hip/hip_runtime.h>
#include <hip/hip_bf16.h>
#include <math.h>

// ---------------------------------------------------------------------------
// Problem constants (reference shapes are fixed)
// ---------------------------------------------------------------------------
constexpr int   Bb = 2, Nn = 2048, Cc = 1024, Hh = 16, Dd = 64;
constexpr float ALPHA = 0.6f;
constexpr float SCALE = 0.125f;          // Dh^-0.5 = 64^-0.5

typedef __bf16 bf16;
typedef __attribute__((ext_vector_type(16))) __bf16 v16bf;
typedef __attribute__((ext_vector_type(8)))  __bf16 v8bf;
typedef __attribute__((ext_vector_type(4)))  __bf16 v4bf;
typedef __attribute__((ext_vector_type(8)))  float  v8f;

// ---------------------------------------------------------------------------
// WMMA fragment helpers (CDNA5 wave32 layouts, CDNA5 ISA §7.12.2)
//   A (16-bit, 16x32): lane m = lane&15, kh = lane>>4
//     elems 0..7  = A[m, k0 + kh*8 .. +7]
//     elems 8..15 = A[m, k0 + 16 + kh*8 .. +7]
//   B (16-bit, 32x16): lane n = lane&15, kh = lane>>4
//     elems 0..7  = B[k0 + kh*8 .. +7, n], elems 8..15 = B[k0 + 16 + kh*8 .. +7, n]   (loaded from row-major W[n, k])
//   C/D (f32 16x16): elem v = D[v + 8*(lane>>4), lane&15]
// ---------------------------------------------------------------------------
__device__ __forceinline__ v8bf ld8(const bf16* p) { return *(const v8bf*)p; }

__device__ __forceinline__ v16bf cat16(v8bf lo, v8bf hi) {
    return __builtin_shufflevector(lo, hi, 0,1,2,3,4,5,6,7,8,9,10,11,12,13,14,15);
}

__device__ __forceinline__ v16bf frag_a(const bf16* row, int k0, int kh) {
    return cat16(ld8(row + k0 + kh * 8), ld8(row + k0 + 16 + kh * 8));
}
__device__ __forceinline__ v16bf frag_b(const bf16* row, int k0, int kh) {
    return cat16(ld8(row + k0 + kh * 8), ld8(row + k0 + 16 + kh * 8));
}

__device__ __forceinline__ v8f wmma_bf16(v16bf a, v16bf b, v8f c) {
    return __builtin_amdgcn_wmma_f32_16x16x32_bf16(false, a, false, b,
                                                   (short)0, c, false, false);
}

// ---------------------------------------------------------------------------
// Software-pipelined 32(M) x 64(N) x K tile: 8 WMMAs per K-step from
// 12 b128 loads, next iteration's fragments loaded before current WMMAs.
// acc layout: acc[r*4 + nt], r = row strip (0..1), nt = col tile (0..3)
// ---------------------------------------------------------------------------
__device__ __forceinline__ void gemm32x64(
    const bf16* __restrict__ arow0, const bf16* __restrict__ arow1,
    const bf16* __restrict__ b0, const bf16* __restrict__ b1,
    const bf16* __restrict__ b2, const bf16* __restrict__ b3,
    int K, int kh, v8f (&acc)[8]) {

    v16bf ac0 = frag_a(arow0, 0, kh);
    v16bf ac1 = frag_a(arow1, 0, kh);
    v16bf bc0 = frag_b(b0, 0, kh), bc1 = frag_b(b1, 0, kh);
    v16bf bc2 = frag_b(b2, 0, kh), bc3 = frag_b(b3, 0, kh);

    for (int k0 = 0; k0 < K; k0 += 32) {
        const int kn = (k0 + 32 < K) ? (k0 + 32) : 0;   // wrap: last prefetch dies
        v16bf an0 = frag_a(arow0, kn, kh);
        v16bf an1 = frag_a(arow1, kn, kh);
        v16bf bn0 = frag_b(b0, kn, kh), bn1 = frag_b(b1, kn, kh);
        v16bf bn2 = frag_b(b2, kn, kh), bn3 = frag_b(b3, kn, kh);

        acc[0] = wmma_bf16(ac0, bc0, acc[0]);
        acc[1] = wmma_bf16(ac0, bc1, acc[1]);
        acc[2] = wmma_bf16(ac0, bc2, acc[2]);
        acc[3] = wmma_bf16(ac0, bc3, acc[3]);
        acc[4] = wmma_bf16(ac1, bc0, acc[4]);
        acc[5] = wmma_bf16(ac1, bc1, acc[5]);
        acc[6] = wmma_bf16(ac1, bc2, acc[6]);
        acc[7] = wmma_bf16(ac1, bc3, acc[7]);

        ac0 = an0; ac1 = an1;
        bc0 = bn0; bc1 = bn1; bc2 = bn2; bc3 = bn3;
    }
}

// ---------------------------------------------------------------------------
// Kernel 0: fp32 -> bf16 cast, vectorized (float4 in, 4x bf16 out)
// ---------------------------------------------------------------------------
__global__ void __launch_bounds__(256) cast_bf16_kernel(const float* __restrict__ in,
                                                        bf16* __restrict__ out, int n4) {
    int i = blockIdx.x * blockDim.x + threadIdx.x;
    int stride = gridDim.x * blockDim.x;
    for (; i < n4; i += stride) {
        float4 f = ((const float4*)in)[i];
        v4bf o;
        o[0] = (bf16)f.x; o[1] = (bf16)f.y; o[2] = (bf16)f.z; o[3] = (bf16)f.w;
        ((v4bf*)out)[i] = o;
    }
}

// ---------------------------------------------------------------------------
// Kernel 1: QKV GEMM.  qkv[4096,3072] = x_bf @ Wqkv_bf^T + bqkv
// Wave owns a 32x64 tile. Epilogue scatters into q (pre-scaled by Dh^-0.5),
// k ([B,H,N,Dh]) and vT ([B,H,Dh,N]) so attention B-fragments are contiguous.
// ---------------------------------------------------------------------------
__global__ void __launch_bounds__(256) qkv_gemm_kernel(
    const bf16* __restrict__ xb, const bf16* __restrict__ wb,
    const float* __restrict__ bias,
    bf16* __restrict__ qb, bf16* __restrict__ kb, bf16* __restrict__ vTb) {

    const int wave = blockIdx.x * 8 + (threadIdx.x >> 5);
    const int lane = threadIdx.x & 31;
    const int tm = wave % (Bb * Nn / 32);       // 0..127 : 32-row strip
    const int tn = wave / (Bb * Nn / 32);       // 0..47  : 64-col strip
    const int m  = lane & 15;
    const int kh = lane >> 4;

    const bf16* arow0 = xb + (size_t)(tm * 32 + m) * Cc;
    const bf16* arow1 = arow0 + (size_t)16 * Cc;
    const bf16* bbase = wb + (size_t)(tn * 64 + m) * Cc;

    v8f acc[8] = {};
    gemm32x64(arow0, arow1,
              bbase, bbase + (size_t)16 * Cc,
              bbase + (size_t)32 * Cc, bbase + (size_t)48 * Cc,
              Cc, kh, acc);

    const int t = tn >> 4;       // 0=q 1=k 2=v  (64-col strips never straddle)
    const int h = tn & 15;
#pragma unroll
    for (int r = 0; r < 2; ++r) {
#pragma unroll
        for (int nt = 0; nt < 4; ++nt) {
            const int d = nt * 16 + m;                 // 0..63 within head
            const float bv = bias[t * Cc + h * Dd + d];
#pragma unroll
            for (int v = 0; v < 8; ++v) {
                const int row = tm * 32 + r * 16 + v + 8 * kh;  // global token row
                const int bidx = row >> 11;                     // /N
                const int n    = row & (Nn - 1);
                const int bh   = bidx * Hh + h;
                const float val = acc[r * 4 + nt][v] + bv;
                if (t == 0)      qb[((size_t)bh * Nn + n) * Dd + d] = (bf16)(val * SCALE);
                else if (t == 1) kb[((size_t)bh * Nn + n) * Dd + d] = (bf16)val;
                else             vTb[((size_t)bh * Dd + d) * Nn + n] = (bf16)val;
            }
        }
    }
}

// ---------------------------------------------------------------------------
// Kernel 2: attention for one (b,h, 16-row strip).
// 4 waves / 128 threads. S strip (16 x 2048 fp32 = 128 KB) lives in LDS.
// ---------------------------------------------------------------------------
__global__ void __launch_bounds__(128) attn_kernel(
    const bf16* __restrict__ qb, const bf16* __restrict__ kb,
    const bf16* __restrict__ vTb, const float* __restrict__ v0,
    bf16* __restrict__ ctx, float* __restrict__ attn_out) {

    extern __shared__ float S[];                 // 16 * 2048 floats
    __shared__ float red[128];
    __shared__ float rowstat[16];

    const int tid  = threadIdx.x;
    const int lane = tid & 31;
    const int wv   = tid >> 5;                   // wave 0..3
    const int bh   = blockIdx.x >> 7;            // (b*H + h)
    const int rt   = blockIdx.x & 127;
    const int n0   = rt * 16;
    const int m    = lane & 15;
    const int kh   = lane >> 4;

    const bf16* qbase = qb  + (size_t)bh * Nn * Dd;
    const bf16* kbase = kb  + (size_t)bh * Nn * Dd;
    const bf16* vbase = vTb + (size_t)bh * Dd * Nn;

    // ---- Phase 1: S = q_scaled @ k^T into LDS (pipelined) ---------------
    {
        const bf16* qrow = qbase + (size_t)(n0 + m) * Dd;
        const v16bf aq0 = frag_a(qrow, 0, kh);
        const v16bf aq1 = frag_a(qrow, 32, kh);

        int ct = wv;
        const bf16* krow = kbase + (size_t)(ct * 16 + m) * Dd;
        v16bf bk0 = frag_b(krow, 0, kh);
        v16bf bk1 = frag_b(krow, 32, kh);

        for (int ci = 0; ci < 32; ++ci) {
            const int ctn = (ci + 1 < 32) ? (ct + 4) : ct;
            const bf16* krn = kbase + (size_t)(ctn * 16 + m) * Dd;
            v16bf bn0 = frag_b(krn, 0, kh);
            v16bf bn1 = frag_b(krn, 32, kh);

            v8f acc = {};
            acc = wmma_bf16(aq0, bk0, acc);
            acc = wmma_bf16(aq1, bk1, acc);
#pragma unroll
            for (int v = 0; v < 8; ++v)
                S[(v + 8 * kh) * Nn + ct * 16 + m] = acc[v];

            ct = ctn; bk0 = bn0; bk1 = bn1;
        }
    }
    __syncthreads();

    // ---- Phase 2: softmax ----------------------------------------------
    {
        const int row = tid & 15, chunk = tid >> 4;     // 8 chunks of 256
        const int base = row * Nn + chunk * 256;
        float pm = -3.0e38f;
        for (int i = 0; i < 256; ++i) pm = fmaxf(pm, S[base + i]);
        red[tid] = pm;
        __syncthreads();
        if (tid < 16) {
            float mx = red[tid];
#pragma unroll
            for (int c = 1; c < 8; ++c) mx = fmaxf(mx, red[tid + 16 * c]);
            rowstat[tid] = mx;
        }
        __syncthreads();
        const float mx = rowstat[row];
        float ps = 0.f;
        for (int i = 0; i < 256; ++i) {
            float e = __expf(S[base + i] - mx);
            S[base + i] = e;
            ps += e;
        }
        __syncthreads();
        red[tid] = ps;
        __syncthreads();
        if (tid < 16) {
            float s = red[tid];
#pragma unroll
            for (int c = 1; c < 8; ++c) s += red[tid + 16 * c];
            rowstat[tid] = 1.f / s;
        }
        __syncthreads();
        // normalize in LDS + coalesced write of attn rows
        float* arow = attn_out + (size_t)bh * Nn * Nn + (size_t)n0 * Nn;
        for (int idx = tid; idx < 16 * Nn; idx += 128) {
            const float p = S[idx] * rowstat[idx >> 11];
            S[idx] = p;
            arow[idx] = p;
        }
        __syncthreads();
    }

    // ---- Phase 3: O = P @ V + ALPHA*(v0 - v) (pipelined) ---------------
    const int dt = wv;                                  // d tile 0..3
    v8f acc = {};
    const float* prow = &S[m * Nn];
    const bf16* vcol = vbase + (size_t)(dt * 16 + m) * Nn;

    v16bf bv = frag_b(vcol, 0, kh);
    for (int m0 = 0; m0 < Nn; m0 += 32) {
        const int mn = (m0 + 32 < Nn) ? (m0 + 32) : 0;
        v16bf bn = frag_b(vcol, mn, kh);

        v16bf a;
#pragma unroll
        for (int i = 0; i < 8; ++i) a[i]     = (bf16)prow[m0 + kh * 8 + i];
#pragma unroll
        for (int i = 0; i < 8; ++i) a[8 + i] = (bf16)prow[m0 + 16 + kh * 8 + i];

        acc = wmma_bf16(a, bv, acc);
        bv = bn;
    }

    const int b_ = bh >> 4, h = bh & 15;
#pragma unroll
    for (int v = 0; v < 8; ++v) {
        const int r = v + 8 * kh;
        const int n = n0 + r;
        const int d = dt * 16 + m;
        const float vv  = (float)vbase[(size_t)d * Nn + n];
        const float v0v = v0[((size_t)bh * Nn + n) * Dd + d];
        const float o = acc[v] + ALPHA * (v0v - vv);
        ctx[((size_t)b_ * Nn + n) * Cc + h * Dd + d] = (bf16)o;
    }
}

// ---------------------------------------------------------------------------
// Kernel 3: output projection. x_out[4096,1024] = ctx_bf @ Wproj_bf^T + bproj
// ---------------------------------------------------------------------------
__global__ void __launch_bounds__(256) proj_gemm_kernel(
    const bf16* __restrict__ ctx, const bf16* __restrict__ wb,
    const float* __restrict__ bias, float* __restrict__ out) {

    const int wave = blockIdx.x * 8 + (threadIdx.x >> 5);
    const int lane = threadIdx.x & 31;
    const int tm = wave % (Bb * Nn / 32);       // 0..127
    const int tn = wave / (Bb * Nn / 32);       // 0..15
    const int m  = lane & 15;
    const int kh = lane >> 4;

    const bf16* arow0 = ctx + (size_t)(tm * 32 + m) * Cc;
    const bf16* arow1 = arow0 + (size_t)16 * Cc;
    const bf16* bbase = wb + (size_t)(tn * 64 + m) * Cc;

    v8f acc[8] = {};
    gemm32x64(arow0, arow1,
              bbase, bbase + (size_t)16 * Cc,
              bbase + (size_t)32 * Cc, bbase + (size_t)48 * Cc,
              Cc, kh, acc);

#pragma unroll
    for (int r = 0; r < 2; ++r) {
#pragma unroll
        for (int nt = 0; nt < 4; ++nt) {
            const int j = tn * 64 + nt * 16 + m;
            const float bvs = bias[j];
#pragma unroll
            for (int v = 0; v < 8; ++v) {
                const int row = tm * 32 + r * 16 + v + 8 * kh;
                out[(size_t)row * Cc + j] = acc[r * 4 + nt][v] + bvs;
            }
        }
    }
}

// ---------------------------------------------------------------------------
// Host-side launcher
// ---------------------------------------------------------------------------
extern "C" void kernel_launch(void* const* d_in, const int* in_sizes, int n_in,
                              void* d_out, int out_size, void* d_ws, size_t ws_size,
                              hipStream_t stream) {
    const float* x     = (const float*)d_in[0];   // [B,N,C]
    const float* v0    = (const float*)d_in[1];   // [B,H,N,Dh]
    const float* Wqkv  = (const float*)d_in[2];   // [3C,C]
    const float* bqkv  = (const float*)d_in[3];   // [3C]
    const float* Wproj = (const float*)d_in[4];   // [C,C]
    const float* bproj = (const float*)d_in[5];   // [C]

    float* x_out = (float*)d_out;                              // B*N*C
    float* attn  = x_out + (size_t)Bb * Nn * Cc;               // B*H*N*N

    // workspace (bf16 elements)
    constexpr size_t SZ_X  = (size_t)Bb * Nn * Cc;    // 4,194,304
    constexpr size_t SZ_WQ = (size_t)3 * Cc * Cc;     // 3,145,728
    constexpr size_t SZ_WP = (size_t)Cc * Cc;         // 1,048,576
    constexpr size_t SZ_T  = (size_t)Bb * Hh * Nn * Dd;

    bf16* xb  = (bf16*)d_ws;
    bf16* wqb = xb  + SZ_X;
    bf16* wpb = wqb + SZ_WQ;
    bf16* qb  = wpb + SZ_WP;
    bf16* kb  = qb  + SZ_T;
    bf16* vTb = kb  + SZ_T;
    bf16* ctx = vTb + SZ_T;

    // 0) casts (vectorized by 4)
    cast_bf16_kernel<<<1024, 256, 0, stream>>>(x,     xb,  (int)(SZ_X  / 4));
    cast_bf16_kernel<<<1024, 256, 0, stream>>>(Wqkv,  wqb, (int)(SZ_WQ / 4));
    cast_bf16_kernel<<<512,  256, 0, stream>>>(Wproj, wpb, (int)(SZ_WP / 4));

    // 1) QKV GEMM: 128 M-strips x 48 N-strips = 6144 waves / 8 per block
    qkv_gemm_kernel<<<768, 256, 0, stream>>>(xb, wqb, bqkv, qb, kb, vTb);

    // 2) attention: B*H*(N/16) = 4096 blocks, 128 threads, 128 KB dynamic LDS
    attn_kernel<<<4096, 128, 16 * Nn * sizeof(float), stream>>>(
        qb, kb, vTb, v0, ctx, attn);

    // 3) projection: 128 x 16 = 2048 waves / 8 per block
    proj_gemm_kernel<<<256, 256, 0, stream>>>(ctx, wpb, bproj, x_out);
}
